// DeepSeekV2MoE_39874476376643
// MI455X (gfx1250) — matrix operand order repaired and checked by reading, not compiled, not run on hardware
//
#include <hip/hip_runtime.h>

// CDNA5 / gfx1250 MoE (DeepSeekV2-style), wave32 + WMMA f16->f32 + TDM.
// HBM-bound: streams 2.2 GB of expert weights once via the Tensor Data Mover
// (double-buffered LDS), WMMA hides all math.

typedef __attribute__((ext_vector_type(16))) _Float16     v16h;
typedef __attribute__((ext_vector_type(8)))  float        v8f;
typedef __attribute__((ext_vector_type(4)))  unsigned int v4u;
typedef __attribute__((ext_vector_type(8)))  int          v8i;
typedef __attribute__((ext_vector_type(4)))  int          v4i;

constexpr int T_    = 128;
constexpr int H_    = 2048;
constexpr int I_    = 1408;
constexpr int E_    = 64;
constexpr int TOPK_ = 6;
constexpr int CAP_  = 128;   // max tokens per expert (= T)

// ------------------------------------------------------------------ TDM issue
// Async-DMA a [rows x 32] f32 tile (row stride = row_stride elems) from global
// memory into LDS at byte offset lds_off. Builds the D# per ISA 8.3/8.4:
//   group0: count=1 | lds_addr | global_addr[56:0] | type=2
//   group1: data_size=4B | tensor_dim0/1 | tile_dim0=32, tile_dim1=rows |
//           tensor_dim0_stride
// Tracked by TENSORcnt; EXEC is ignored (issue from a single wave).
__device__ __forceinline__ void tdm_load_2d(unsigned lds_off,
                                            const float* gptr,
                                            int rows, int row_stride) {
  unsigned long long ga = (unsigned long long)(size_t)gptr;
  v4u g0;
  g0.x = 1u;                                            // count=1, user D#
  g0.y = lds_off;                                       // LDS byte address
  g0.z = (unsigned)(ga & 0xFFFFFFFFu);                  // global_addr[31:0]
  g0.w = (unsigned)((ga >> 32) & 0x01FFFFFFu)           // global_addr[56:32]
         | (2u << 30);                                  // type=2 ("image")
  unsigned dim0   = (unsigned)row_stride;               // tensor extent x
  unsigned dim1   = (unsigned)rows;                     // tensor extent y
  unsigned tile0  = 32u;                                // tile x (elems)
  unsigned tile1  = (unsigned)rows;                     // tile y (rows)
  unsigned long long s0 = (unsigned long long)row_stride;  // dim0 stride
  v8i g1;
  g1[0] = (int)(2u << 16);                              // data_size=4B; mask=0
  g1[1] = (int)((dim0 & 0xFFFFu) << 16);                // tensor_dim0[15:0]
  g1[2] = (int)((dim0 >> 16) | ((dim1 & 0xFFFFu) << 16));
  g1[3] = (int)((dim1 >> 16) | (tile0 << 16));          // tile_dim0
  g1[4] = (int)tile1;                                   // tile_dim1 (dim2=0)
  g1[5] = (int)(s0 & 0xFFFFFFFFull);                    // dim0_stride[31:0]
  g1[6] = (int)((s0 >> 32) & 0xFFFFull);                // dim0_stride[47:32]
  g1[7] = 0;
  v4i z4 = {0, 0, 0, 0};
#if defined(__clang_major__) && (__clang_major__ >= 23)
  v8i z8 = {0, 0, 0, 0, 0, 0, 0, 0};
  __builtin_amdgcn_tensor_load_to_lds(g0, g1, z4, z4, z8, 0);
#else
  __builtin_amdgcn_tensor_load_to_lds(g0, g1, z4, z4, 0);
#endif
}

// ---------------------------------------------------------------- utilities
__global__ void zero_f32_kernel(float* p, int n) {
  int i = blockIdx.x * blockDim.x + threadIdx.x;
  if (i < n) p[i] = 0.f;
}

__global__ void cvt_f32_f16_kernel(const float* __restrict__ src,
                                   _Float16* __restrict__ dst, int n) {
  int i = blockIdx.x * blockDim.x + threadIdx.x;
  if (i < n) dst[i] = (_Float16)src[i];
}

// ---------------------------------------------------------------- router
__global__ __launch_bounds__(64)
void router_kernel(const float* __restrict__ x, const float* __restrict__ gw,
                   float* __restrict__ combine) {
  __shared__ float sl[E_];
  __shared__ int   ssel[E_];
  int t = blockIdx.x;
  int e = threadIdx.x;
  const float* xr = x  + (size_t)t * H_;
  const float* gr = gw + (size_t)e * H_;
  float acc = 0.f;
  for (int h = 0; h < H_; h += 4) {
    float4 xv = *(const float4*)(xr + h);
    float4 gv = *(const float4*)(gr + h);
    acc += xv.x * gv.x + xv.y * gv.y + xv.z * gv.z + xv.w * gv.w;
  }
  sl[e] = acc;
  ssel[e] = 0;
  __syncthreads();
  if (e == 0) {
    float mx = sl[0];
    for (int i = 1; i < E_; ++i) mx = fmaxf(mx, sl[i]);
    float sum = 0.f;
    for (int i = 0; i < E_; ++i) { float p = __expf(sl[i] - mx); sl[i] = p; sum += p; }
    float inv = 1.f / sum;
    for (int k = 0; k < TOPK_; ++k) {
      int best = 0; float bv = -1.f;
      for (int i = 0; i < E_; ++i)
        if (!ssel[i] && sl[i] > bv) { bv = sl[i]; best = i; }
      ssel[best] = 1;
    }
    for (int i = 0; i < E_; ++i)
      combine[(size_t)t * E_ + i] = ssel[i] ? sl[i] * inv : 0.f;
  }
}

// ---------------------------------------------------------------- token lists
__global__ __launch_bounds__(32)
void build_lists_kernel(const float* __restrict__ combine, int* __restrict__ cnt,
                        int* __restrict__ tok, float* __restrict__ wgt) {
  int e = blockIdx.x;
  if (threadIdx.x != 0) return;
  int c = 0;
  for (int t = 0; t < T_; ++t) {
    float w = combine[(size_t)t * E_ + e];
    if (w > 0.f) { tok[e * CAP_ + c] = t; wgt[e * CAP_ + c] = w; ++c; }
  }
  cnt[e] = c;
}

// ---------------------------------------------------------------- gate/up + SwiGLU
// Block: 128 threads (4 waves). Tile: 16 tokens x 64 intermediate cols.
// Weights DMA'd f32 into double-buffered LDS by the TDM; converted to f16 at
// B-fragment read (each element consumed exactly once).
__global__ __launch_bounds__(128)
void moe_gateup_kernel(const _Float16* __restrict__ xh,
                       const float* __restrict__ w1g,
                       const float* __restrict__ w1u,
                       const int* __restrict__ cnt,
                       const int* __restrict__ tok,
                       _Float16* __restrict__ a_buf) {
  int e  = blockIdx.z;
  int m0 = blockIdx.y * 16;
  int ne = cnt[e];
  if (m0 >= ne) return;                  // sparse early-exit
  int i0 = blockIdx.x * 64;

  __shared__ float lgf[2][64][32];       // w1_gate tiles (f32, double buffered)
  __shared__ float luf[2][64][32];       // w1_up tiles
  __shared__ int toks[16];

  int tid  = threadIdx.x;
  int lane = tid & 31;
  int wave = tid >> 5;

  if (tid < 16) {
    int m = m0 + tid;
    toks[tid] = tok[e * CAP_ + (m < ne ? m : ne - 1)];
  }

  const float* wg_base = w1g + ((size_t)e * I_ + i0) * H_;
  const float* wu_base = w1u + ((size_t)e * I_ + i0) * H_;

  if (tid == 0) {                        // prime the TDM pipeline (chunk 0)
    tdm_load_2d((unsigned)(size_t)&lgf[0][0][0], wg_base, 64, H_);
    tdm_load_2d((unsigned)(size_t)&luf[0][0][0], wu_base, 64, H_);
  }
  __syncthreads();                       // publish toks[]

  const _Float16* arow = xh + (size_t)toks[lane & 15] * H_;
  int kbase = (lane >> 4) * 16;          // A frag: K half select
  int nloc  = wave * 16 + (lane & 15);   // B frag: local column (i) index
  int bk    = (lane >> 4) * 16;          // B frag: K half select

  v8f cg = {};
  v8f cu = {};

  for (int k0 = 0, c = 0; k0 < H_; k0 += 32, ++c) {
    int p = c & 1;
    if (tid == 0) {
      if (k0 + 32 < H_) {                // DMA next chunk into the other buffer
        tdm_load_2d((unsigned)(size_t)&lgf[p ^ 1][0][0], wg_base + k0 + 32, 64, H_);
        tdm_load_2d((unsigned)(size_t)&luf[p ^ 1][0][0], wu_base + k0 + 32, 64, H_);
        __builtin_amdgcn_s_wait_tensorcnt((short)2);  // current chunk landed
      } else {
        __builtin_amdgcn_s_wait_tensorcnt((short)0);
      }
    }
    __syncthreads();

    // A fragment (16x32 f16): lane M = L%16; K = (L/16)*16+j
    v16h a;
#pragma unroll
    for (int j = 0; j < 8; ++j) a[j] = arow[k0 + kbase + j];
#pragma unroll
    for (int j = 0; j < 8; ++j) a[8 + j] = arow[k0 + 8 + kbase + j];

    // B fragments (32x16 f16): lane N = L%16; K = (L/16)*16+j; cvt on read
    v16h bg, bu;
#pragma unroll
    for (int j = 0; j < 16; ++j) {
      bg[j] = (_Float16)lgf[p][nloc][bk + j];
      bu[j] = (_Float16)luf[p][nloc][bk + j];
    }

    cg = __builtin_amdgcn_wmma_f32_16x16x32_f16(false, a, false, bg, (short)0, cg,
                                                false, false);
    cu = __builtin_amdgcn_wmma_f32_16x16x32_f16(false, a, false, bu, (short)0, cu,
                                                false, false);
    __syncthreads();                     // buffer p free for DMA reuse
  }

  // epilogue: a = silu(h) * u, store f16 activations
  int ncol = i0 + wave * 16 + (lane & 15);
  size_t rbase = (size_t)e * CAP_ + m0;
#pragma unroll
  for (int v = 0; v < 8; ++v) {
    int m = (lane >> 4) * 8 + v;         // C layout: VGPR v -> M=(L/16)*8+v
    float hv = cg[v];
    float s  = hv * __builtin_amdgcn_rcpf(1.f + __expf(-hv));
    a_buf[(rbase + m) * I_ + ncol] = (_Float16)(s * cu[v]);
  }
}

// ---------------------------------------------------------------- down proj + combine
__global__ __launch_bounds__(128)
void moe_down_kernel(const _Float16* __restrict__ a_buf,
                     const float* __restrict__ w2,
                     const int* __restrict__ cnt,
                     const int* __restrict__ tok,
                     const float* __restrict__ wgt,
                     float* __restrict__ y) {
  int e  = blockIdx.z;
  int m0 = blockIdx.y * 16;
  int ne = cnt[e];
  if (m0 >= ne) return;
  int h0 = blockIdx.x * 64;

  __shared__ float lwf[2][64][32];       // w2 tiles (f32, double buffered)

  int tid  = threadIdx.x;
  int lane = tid & 31;
  int wave = tid >> 5;

  const _Float16* arow =
      a_buf + ((size_t)e * CAP_ + m0 + (lane & 15)) * I_;
  int kbase = (lane >> 4) * 16;
  int nloc  = wave * 16 + (lane & 15);
  int bk    = (lane >> 4) * 16;

  const float* w2_base = w2 + ((size_t)e * H_ + h0) * I_;

  if (tid == 0)
    tdm_load_2d((unsigned)(size_t)&lwf[0][0][0], w2_base, 64, I_);

  v8f c = {};

  for (int k0 = 0, cidx = 0; k0 < I_; k0 += 32, ++cidx) {
    int p = cidx & 1;
    if (tid == 0) {
      if (k0 + 32 < I_) {
        tdm_load_2d((unsigned)(size_t)&lwf[p ^ 1][0][0], w2_base + k0 + 32, 64, I_);
        __builtin_amdgcn_s_wait_tensorcnt((short)1);
      } else {
        __builtin_amdgcn_s_wait_tensorcnt((short)0);
      }
    }
    __syncthreads();

    v16h a;
#pragma unroll
    for (int j = 0; j < 8; ++j) a[j] = arow[k0 + kbase + j];
#pragma unroll
    for (int j = 0; j < 8; ++j) a[8 + j] = arow[k0 + 8 + kbase + j];

    v16h b;
#pragma unroll
    for (int j = 0; j < 16; ++j) b[j] = (_Float16)lwf[p][nloc][bk + j];

    c = __builtin_amdgcn_wmma_f32_16x16x32_f16(false, a, false, b, (short)0, c,
                                               false, false);
    __syncthreads();
  }

  int hcol = h0 + wave * 16 + (lane & 15);
#pragma unroll
  for (int v = 0; v < 8; ++v) {
    int m = (lane >> 4) * 8 + v;
    if (m0 + m < ne) {
      int   t  = tok[e * CAP_ + m0 + m];
      float cw = wgt[e * CAP_ + m0 + m];
      atomicAdd(&y[(size_t)t * H_ + hcol], cw * c[v]);
    }
  }
}

// ---------------------------------------------------------------- launcher
extern "C" void kernel_launch(void* const* d_in, const int* in_sizes, int n_in,
                              void* d_out, int out_size, void* d_ws, size_t ws_size,
                              hipStream_t stream) {
  const float* x   = (const float*)d_in[0];   // [T,H]
  const float* gw  = (const float*)d_in[1];   // [E,H]
  const float* w1g = (const float*)d_in[2];   // [E,I,H]
  const float* w1u = (const float*)d_in[3];   // [E,I,H]
  const float* w2  = (const float*)d_in[4];   // [E,H,I]
  float* y = (float*)d_out;                   // [T,H]

  char* ws = (char*)d_ws;
  float*    combine = (float*)   (ws + 0);        //  32768 B
  int*      cnt     = (int*)     (ws + 32768);    //    256 B
  int*      tok     = (int*)     (ws + 33024);    //  32768 B
  float*    wgt     = (float*)   (ws + 65792);    //  32768 B
  _Float16* xh      = (_Float16*)(ws + 98560);    // 524288 B
  _Float16* a_buf   = (_Float16*)(ws + 622848);   // 64*128*1408*2 = 23068672 B

  int nxy = T_ * H_;  // 262144
  zero_f32_kernel<<<(nxy + 255) / 256, 256, 0, stream>>>(y, nxy);
  cvt_f32_f16_kernel<<<(nxy + 255) / 256, 256, 0, stream>>>(x, xh, nxy);
  router_kernel<<<T_, 64, 0, stream>>>(x, gw, combine);
  build_lists_kernel<<<E_, 32, 0, stream>>>(combine, cnt, tok, wgt);

  dim3 g1(I_ / 64, CAP_ / 16, E_);   // (22, 8, 64) — most m-tiles early-exit
  moe_gateup_kernel<<<g1, 128, 0, stream>>>(xh, w1g, w1u, cnt, tok, a_buf);

  dim3 g2(H_ / 64, CAP_ / 16, E_);   // (32, 8, 64)
  moe_down_kernel<<<g2, 128, 0, stream>>>(a_buf, w2, cnt, tok, wgt, y);
}
